// LSTMModel_34153579938497
// MI455X (gfx1250) — matrix operand order repaired and checked by reading, not compiled, not run on hardware
//
#include <hip/hip_runtime.h>
#include <hip/hip_bf16.h>
#include <math.h>

#define B_   64
#define T_   512
#define D_   1024
#define H_   1024
#define KTOT 2048          // D + H (concatenated GEMM K)
#define NG   4096          // 4*H gate rows

typedef __attribute__((ext_vector_type(16))) __bf16 v16bf;
typedef __attribute__((ext_vector_type(8)))  float  v8f;

union V16U { v16bf v; struct { uint4 lo; uint4 hi; } q; };

// Assemble a v16bf WMMA operand from two 16-byte chunks (elements 0..7 -> VGPR0..3,
// elements 8..15 -> VGPR4..7 per the CDNA5 16-bit A/B register layout).
__device__ inline v16bf ld_pair(const __bf16* p0, const __bf16* p1) {
  V16U u;
  u.q.lo = *(const uint4*)p0;
  u.q.hi = *(const uint4*)p1;
  return u.v;
}

// ---------------------------------------------------------------------------
// Pre-pass: pack W = [Wx | Wh] (4096 x 2048) to bf16, and bsum = bx + bh.
// Runs once per call; the 512 recurrent steps then stream bf16 from L2.
// ---------------------------------------------------------------------------
__global__ void __launch_bounds__(256)
pack_weights(const float* __restrict__ Wx, const float* __restrict__ Wh,
             const float* __restrict__ bx, const float* __restrict__ bh,
             __bf16* __restrict__ Wc, float* __restrict__ bsum) {
  size_t stride = (size_t)gridDim.x * blockDim.x;
  for (size_t i = (size_t)blockIdx.x * blockDim.x + threadIdx.x;
       i < (size_t)NG * KTOT; i += stride) {
    size_t n = i >> 11;
    int    k = (int)(i & (KTOT - 1));
    float  v = (k < D_) ? Wx[n * D_ + k] : Wh[n * H_ + (k - D_)];
    Wc[i] = (__bf16)v;
  }
  for (size_t i = (size_t)blockIdx.x * blockDim.x + threadIdx.x; i < NG; i += stride)
    bsum[i] = bx[i] + bh[i];
}

__global__ void __launch_bounds__(256)
init_state(float* __restrict__ h0, float* __restrict__ c) {
  int i = blockIdx.x * blockDim.x + threadIdx.x;
  if (i < B_ * H_) { h0[i] = 0.0f; c[i] = 0.0f; }
}

// ---------------------------------------------------------------------------
// One recurrent step. grid = (H/16, B/16), block = 128 (4 waves = 4 gates).
// Each block owns a 16x16 (batch x hidden) patch: computes i,f,g,o tiles with
// WMMA over K=2048, then does the LSTM cell update locally.
// ---------------------------------------------------------------------------
__global__ void __launch_bounds__(128)
lstm_step(const float* __restrict__ x, const float* __restrict__ tm,
          const __bf16* __restrict__ Wc, const float* __restrict__ bsum,
          const float* __restrict__ h_in, float* __restrict__ h_out,
          float* __restrict__ c, float* __restrict__ out, int t)
{
  __shared__ __align__(16) unsigned char smem[16 * KTOT * 2];   // 64 KB
  __bf16* As = (__bf16*)smem;          // A tile: [new_x | h], 16 x 2048 bf16

  const int nt  = blockIdx.x;          // hidden tile 0..63
  const int mt  = blockIdx.y;          // batch tile 0..3
  const int tid = threadIdx.x;

  // ---- Stage A into LDS (teacher-mask blend + bf16 convert, done once) ----
  for (int i = tid; i < 16 * KTOT; i += 128) {
    int b  = i >> 11;
    int k  = i & (KTOT - 1);
    int bg = mt * 16 + b;
    float v;
    if (k < D_) {
      float m = tm[(size_t)bg * T_ + t];
      v = m * h_in[(size_t)bg * H_ + k]
        + (1.0f - m) * x[((size_t)bg * T_ + t) * D_ + k];
    } else {
      v = h_in[(size_t)bg * H_ + (k - D_)];
    }
    As[i] = (__bf16)v;
  }
  __syncthreads();

  // ---- WMMA GEMM: wave w computes gate w's 16x16 tile over K=2048 ----
  const int wave = tid >> 5;           // gate: 0=i 1=f 2=g 3=o
  const int lane = tid & 31;
  const int l16  = lane & 15;
  const int half = lane >> 4;

  const __bf16* wrow = Wc + ((size_t)wave * H_ + (size_t)nt * 16 + l16) * KTOT;
  const __bf16* arow = As + l16 * KTOT;

  v8f acc = {};
  #pragma unroll 8
  for (int kb = 0; kb < KTOT / 32; ++kb) {
    // A (16-bit 16x32): lane K pattern = half*8 + {0..7} and half*8 + {16..23}
    int kA = kb * 32 + half * 8;
    v16bf a = ld_pair(arow + kA, arow + kA + 16);
    // B (16-bit 32x16): lane holds one output row, K = half*8 + {0..7} and half*8 + {16..23}
    int kB = kb * 32 + half * 8;
    v16bf bm = ld_pair(wrow + kB, wrow + kB + 16);
    acc = __builtin_amdgcn_wmma_f32_16x16x32_bf16(
        /*neg_a=*/false, a, /*neg_b=*/false, bm,
        /*c_mod=*/(short)0, acc, /*reuse_a=*/false, /*reuse_b=*/false);
  }
  __syncthreads();                     // all waves done reading As

  // Spill the 4 gate accumulator tiles to LDS (reuse As space): C/D layout
  // lane l, vgpr p -> row M = p + 8*(l>>4), col N = l&15.
  float* Gs = (float*)smem;            // [4][16][16]
  #pragma unroll
  for (int p = 0; p < 8; ++p)
    Gs[wave * 256 + (p + 8 * half) * 16 + l16] = acc[p];
  __syncthreads();

  // ---- LSTM cell update for this 16x16 state patch ----
  for (int e = tid; e < 256; e += 128) {
    int b  = e >> 4, j = e & 15;
    int bg = mt * 16 + b;
    int jg = nt * 16 + j;
    float iv = Gs[0 * 256 + e] + bsum[0 * H_ + jg];
    float fv = Gs[1 * 256 + e] + bsum[1 * H_ + jg];
    float gv = Gs[2 * 256 + e] + bsum[2 * H_ + jg];
    float ov = Gs[3 * 256 + e] + bsum[3 * H_ + jg];
    float si = 1.0f / (1.0f + __expf(-iv));
    float sf = 1.0f / (1.0f + __expf(-fv));
    float tg = tanhf(gv);
    float so = 1.0f / (1.0f + __expf(-ov));
    size_t idx = (size_t)bg * H_ + jg;
    float cy = c[idx] * sf + si * tg;
    float hy = so * tanhf(cy);
    c[idx]     = cy;
    h_out[idx] = hy;
    out[((size_t)bg * T_ + t) * H_ + jg] = hy;
  }
}

__global__ void __launch_bounds__(256)
finalize(const float* __restrict__ hf, const float* __restrict__ c,
         float* __restrict__ out_tail) {
  int i = blockIdx.x * blockDim.x + threadIdx.x;
  if (i < B_ * H_) {
    out_tail[i]            = hf[i];
    out_tail[B_ * H_ + i]  = c[i];
  }
}

// ---------------------------------------------------------------------------
extern "C" void kernel_launch(void* const* d_in, const int* in_sizes, int n_in,
                              void* d_out, int out_size, void* d_ws, size_t ws_size,
                              hipStream_t stream) {
  const float* x  = (const float*)d_in[0];   // [B,T,D]
  const float* tm = (const float*)d_in[1];   // [B,T]
  const float* Wx = (const float*)d_in[2];   // [4H,D]
  const float* bx = (const float*)d_in[3];   // [4H]
  const float* Wh = (const float*)d_in[4];   // [4H,H]
  const float* bh = (const float*)d_in[5];   // [4H]
  float* out = (float*)d_out;                // [B,T,H] ++ hn[B,H] ++ cn[B,H]

  char* ws = (char*)d_ws;
  // ws layout (256-aligned): bf16 weights 16MB | bsum 16KB | h0 | h1 | c
  __bf16* Wc   = (__bf16*)(ws);
  float*  bsum = (float*)(ws + 16777216);
  float*  h0   = (float*)(ws + 16793600);
  float*  h1   = (float*)(ws + 16793600 + 262144);
  float*  cst  = (float*)(ws + 16793600 + 524288);

  pack_weights<<<4096, 256, 0, stream>>>(Wx, Wh, bx, bh, Wc, bsum);
  init_state<<<(B_ * H_ + 255) / 256, 256, 0, stream>>>(h0, cst);

  for (int t = 0; t < T_; ++t) {
    const float* hin  = (t & 1) ? h1 : h0;
    float*       hout = (t & 1) ? h0 : h1;
    lstm_step<<<dim3(H_ / 16, B_ / 16), 128, 0, stream>>>(
        x, tm, Wc, bsum, hin, hout, cst, out, t);
  }
  // T=512 even -> final h landed in h0
  finalize<<<(B_ * H_ + 255) / 256, 256, 0, stream>>>(
      h0, cst, out + (size_t)B_ * T_ * H_);
}
